// CustomConv2d_76063870812450
// MI455X (gfx1250) — matrix operand order repaired and checked by reading, not compiled, not run on hardware
//
#include <hip/hip_runtime.h>

// Implicit-GEMM 3x3 conv for MI455X (gfx1250, wave32, WMMA).
// M = 32*56*56 = 100352 pixels, N = 256 out-channels, K = 9*128 = 1152
// (K reordered as k = rs*128 + c so the im2col unpack needs no divisions).
// 59.2 GFLOP vs 155 MB -> matrix-pipe bound; f16 WMMA w/ f32 accumulate.

typedef __attribute__((ext_vector_type(16))) _Float16 v16h;
typedef __attribute__((ext_vector_type(8)))  float    v8f;

#define TILE_M 128
#define TILE_N 128
#define TILE_K 32
#define KDIM   1152
#define HW     3136      // 56*56
#define CHW    401408    // 128*3136
#define LDSS   34        // padded LDS row stride in f16 (17 banks, conflict-free)

// ---- one-time weight prep: w[OIHW] f32 -> wB[n][rs][c] f16 in workspace ----
__global__ __launch_bounds__(256)
void wprep_kernel(const float* __restrict__ w, _Float16* __restrict__ wB)
{
    int idx = blockIdx.x * 256 + threadIdx.x;     // 256*1152 = 294912 elements
    if (idx < 256 * KDIM) {
        int n  = idx / KDIM;
        int t  = idx - n * KDIM;                  // t = rs*128 + c
        int rs = t >> 7;
        int c  = t & 127;
        wB[idx] = (_Float16)w[n * KDIM + c * 9 + rs];
    }
}

__global__ __launch_bounds__(256)
void conv3x3_wmma_kernel(const float* __restrict__ x,
                         const _Float16* __restrict__ wB,
                         const float* __restrict__ bias,
                         float* __restrict__ out)
{
    __shared__ _Float16 lA[TILE_M * LDSS];  // 8704 B  im2col activations [m][c]
    __shared__ _Float16 lB[TILE_N * LDSS];  // 8704 B  weights            [n][c]

    const int tid    = threadIdx.x;
    const int lane   = tid & 31;
    const int wave   = tid >> 5;     // 0..7
    const int wave_m = wave >> 1;    // 0..3 -> 32 rows each
    const int wave_n = wave & 1;     // 0..1 -> 64 cols each

    const int m_block = blockIdx.x * TILE_M;   // 784 blocks
    const int n_block = blockIdx.y * TILE_N;   // 2 blocks

    // ---- per-thread staging coordinates ----
    const int a_p0  = tid >> 5;          // A: k-pair base (0..7), pairs p, p+8
    const int a_ml0 = tid & 31;          // A: m_local base, 4 groups step 32
    int a_nimg[4], a_oh[4], a_ow[4];
    #pragma unroll
    for (int ii = 0; ii < 4; ++ii) {
        int m = m_block + a_ml0 + ii * 32;
        a_nimg[ii] = m / HW;
        int hw = m - a_nimg[ii] * HW;
        a_oh[ii] = hw / 56;
        a_ow[ii] = hw - a_oh[ii] * 56;
    }
    const int b_c2 = (tid & 15) * 2;     // B: c-pair offset within chunk
    const int b_n0 = tid >> 4;           // B: n_local base (0..15), 8 groups

    const int half = lane >> 4;
    const int l15  = lane & 15;

    v8f acc[2][4] = {};

    for (int rs = 0; rs < 9; ++rs) {
        const int r = rs / 3;            // uniform (SALU), 9x total
        const int s = rs - r * 3;

        // Per-rs im2col bases + padding masks (branch-free thereafter).
        int   a_base[4];
        float a_msk[4];
        #pragma unroll
        for (int ii = 0; ii < 4; ++ii) {
            int ih = a_oh[ii] + r - 1;
            int iw = a_ow[ii] + s - 1;
            bool ok = ((unsigned)ih < 56u) & ((unsigned)iw < 56u);
            a_base[ii] = ok ? (a_nimg[ii] * CHW + ih * 56 + iw) : 0;
            a_msk[ii]  = ok ? 1.0f : 0.0f;
        }

        for (int cb = 0; cb < 128; cb += TILE_K) {
            // ---- stage A: 128 m x 32 c, packed f16 pairs, unconditional loads ----
            #pragma unroll
            for (int jj = 0; jj < 2; ++jj) {
                int p  = a_p0 + jj * 8;          // pair index 0..15
                int c0 = cb + 2 * p;
                #pragma unroll
                for (int ii = 0; ii < 4; ++ii) {
                    float v0 = x[a_base[ii] + (c0    ) * HW] * a_msk[ii];
                    float v1 = x[a_base[ii] + (c0 + 1) * HW] * a_msk[ii];
                    union { _Float16 h[2]; unsigned u; } pk;
                    pk.h[0] = (_Float16)v0;
                    pk.h[1] = (_Float16)v1;
                    *(unsigned*)&lA[(a_ml0 + ii * 32) * LDSS + 2 * p] = pk.u;
                }
            }
            // ---- stage B: 128 n x 32 c, already f16 [n][rs][c] contiguous ----
            #pragma unroll
            for (int jj = 0; jj < 8; ++jj) {
                int nl = b_n0 + jj * 16;
                unsigned v = *(const unsigned*)
                    &wB[((n_block + nl) * 9 + rs) * 128 + cb + b_c2];
                *(unsigned*)&lB[nl * LDSS + b_c2] = v;
            }

            __syncthreads();

            // ---- WMMA fragments from LDS (ISA 7.12.2 wave32 layouts) ----
            union { v16h h; unsigned u[8]; } afrag[2], bfrag[4];
            #pragma unroll
            for (int mi = 0; mi < 2; ++mi) {
                const _Float16* base = &lA[(wave_m * 32 + mi * 16 + l15) * LDSS];
                #pragma unroll
                for (int i = 0; i < 8; ++i) {
                    // A 16x32: VGPR i<4 -> K = 2i + 8*half ; i>=4 -> 16 + 2(i-4) + 8*half
                    int K0 = (i < 4) ? (2 * i + 8 * half)
                                     : (16 + 2 * (i - 4) + 8 * half);
                    afrag[mi].u[i] = *(const unsigned*)(base + K0);
                }
            }
            #pragma unroll
            for (int ni = 0; ni < 4; ++ni) {
                const _Float16* base = &lB[(wave_n * 64 + ni * 16 + l15) * LDSS];
                #pragma unroll
                for (int v = 0; v < 8; ++v) {
                    // B 32x16: lane = column N, VGPR v<4 -> K = 2v + 8*half ; v>=4 -> 16 + 2(v-4) + 8*half
                    int K0 = (v < 4) ? (2 * v + 8 * half)
                                     : (16 + 2 * (v - 4) + 8 * half);
                    bfrag[ni].u[v] = *(const unsigned*)(base + K0);
                }
            }

            // ---- 8 WMMAs per wave per K-step ----
            #pragma unroll
            for (int mi = 0; mi < 2; ++mi)
                #pragma unroll
                for (int ni = 0; ni < 4; ++ni)
                    acc[mi][ni] = __builtin_amdgcn_wmma_f32_16x16x32_f16(
                        false, afrag[mi].h, false, bfrag[ni].h,
                        (short)0, acc[mi][ni], false, false);

            __syncthreads();
        }
    }

    // ---- epilogue: bias add + b128 stores ----
    // C/D layout: lane l15 = column N, VGPR r = row (r + 8*half); each lane's
    // 8 rows are memory-contiguous (3136 % 8 == 0, base % 8 == 0).
    #pragma unroll
    for (int mi = 0; mi < 2; ++mi) {
        int m0   = m_block + wave_m * 32 + mi * 16 + 8 * half;
        int nimg = m0 / HW;
        int rem  = m0 - nimg * HW;
        #pragma unroll
        for (int ni = 0; ni < 4; ++ni) {
            int ko   = n_block + wave_n * 64 + ni * 16 + l15;
            float bv = bias[ko];
            float* p = out + ((size_t)(nimg * 256 + ko)) * HW + rem;
            v8f a = acc[mi][ni];
            float4 lo = make_float4(a[0] + bv, a[1] + bv, a[2] + bv, a[3] + bv);
            float4 hi = make_float4(a[4] + bv, a[5] + bv, a[6] + bv, a[7] + bv);
            *(float4*)(p)     = lo;
            *(float4*)(p + 4) = hi;
        }
    }
}

extern "C" void kernel_launch(void* const* d_in, const int* in_sizes, int n_in,
                              void* d_out, int out_size, void* d_ws, size_t ws_size,
                              hipStream_t stream) {
    const float* x    = (const float*)d_in[0];   // [32,128,56,56]
    const float* w    = (const float*)d_in[1];   // [256,128,3,3]
    const float* bias = (const float*)d_in[2];   // [256]
    float* out        = (float*)d_out;           // [32,256,56,56]

    _Float16* wB = (_Float16*)d_ws;              // 256*1152*2 = 576 KB scratch

    wprep_kernel<<<(256 * KDIM + 255) / 256, 256, 0, stream>>>(w, wB);

    dim3 grid(100352 / TILE_M, 256 / TILE_N);    // (784, 2)
    dim3 block(256);
    conv3x3_wmma_kernel<<<grid, block, 0, stream>>>(x, wB, bias, out);
}
